// SwinTransformer_51238959841389
// MI455X (gfx1250) — matrix operand order repaired and checked by reading, not compiled, not run on hardware
//
#include <hip/hip_runtime.h>
#include <hip/hip_bf16.h>

// ---------------- constants (all shapes compile-time for this problem) ------
#define WSZ   7
#define SSZ   3
#define NHEAD 12
#define HD    32
#define IMG   112
#define NWS   16          // windows per side = 112/7
#define CDIM  384
#define CQKV  1152
#define NTOK  49          // tokens per window
#define BW    2048        // 8 * 16*16 windows
#define KT    12          // K tiles of 32 (384/32)

typedef __attribute__((ext_vector_type(4)))  __bf16 bf16x4;
typedef __attribute__((ext_vector_type(8)))  __bf16 bf16x8;
typedef __attribute__((ext_vector_type(16))) __bf16 bf16x16;
typedef __attribute__((ext_vector_type(8)))  float  f32x8;

__device__ __forceinline__ f32x8 zero8() {
  f32x8 v;
#pragma unroll
  for (int i = 0; i < 8; ++i) v[i] = 0.0f;
  return v;
}

__device__ __forceinline__ f32x8 wmma_bf16(bf16x16 a, bf16x16 b, f32x8 c) {
  return __builtin_amdgcn_wmma_f32_16x16x32_bf16(
      false, a, false, b, (short)0, c, false, false);
}

// A fragment (16-bit A 16x32, ISA 7.12.2): lane&15 = row, half = lane>>4.
__device__ __forceinline__ bf16x16 fragA(const __bf16* rowChunk, int half) {
  bf16x8 lo = *(const bf16x8*)(rowChunk + 8 * half);
  bf16x8 hi = *(const bf16x8*)(rowChunk + 16 + 8 * half);
  bf16x16 r;
#pragma unroll
  for (int i = 0; i < 8; ++i) { r[i] = lo[i]; r[i + 8] = hi[i]; }
  return r;
}

// B fragment (16-bit B 32x16): lane&15 = column; colChunk -> 32 contiguous K.
__device__ __forceinline__ bf16x16 fragB(const __bf16* colChunk, int half) {
  bf16x8 lo = *(const bf16x8*)(colChunk + 8 * half);
  bf16x8 hi = *(const bf16x8*)(colChunk + 16 + 8 * half);
  bf16x16 r;
#pragma unroll
  for (int i = 0; i < 8; ++i) { r[i] = lo[i]; r[i + 8] = hi[i]; }
  return r;
}

// CDNA5 async global->LDS copy (ASYNCcnt-tracked), 16B per lane.
__device__ __forceinline__ void async_ld16(unsigned lds_off, const void* g) {
  asm volatile("global_load_async_to_lds_b128 %0, %1, off"
               :: "v"(lds_off), "v"(g) : "memory");
}
__device__ __forceinline__ void wait_async0() {
  asm volatile("s_wait_asynccnt 0x0" ::: "memory");
}

// shifted-window gather/scatter position (bijection; same both directions)
__device__ __forceinline__ int winRowBase(int w, int r) {
  int b = w >> 8, wi = w & 255, wh = wi >> 4, ww = wi & 15;
  int i = r / WSZ, j = r % WSZ;
  int gh = (wh * WSZ + i + SSZ) % IMG;
  int gw = (ww * WSZ + j + SSZ) % IMG;
  return ((b * IMG * IMG) + gh * IMG + gw) * CDIM;
}

// ---------------- kernel 0: weights f32 -> bf16 fragment-order layout -------
// layout: [ntile][ktile][col(0..15)][k(0..31)]
__global__ __launch_bounds__(256) void prep_weights(
    const float* __restrict__ qkv_w, const float* __restrict__ proj_w,
    __bf16* __restrict__ pwQ, __bf16* __restrict__ pwP) {
  int idx = blockIdx.x * 256 + threadIdx.x;
  const int NQ = CDIM * CQKV;
  if (idx < NQ) {
    int k = idx / CQKV, n = idx % CQKV;
    pwQ[(((n >> 4) * KT + (k >> 5)) * 16 + (n & 15)) * 32 + (k & 31)] =
        (__bf16)qkv_w[idx];
  } else if (idx < NQ + CDIM * CDIM) {
    int j = idx - NQ;
    int k = j / CDIM, n = j % CDIM;
    pwP[(((n >> 4) * KT + (k >> 5)) * 16 + (n & 15)) * 32 + (k & 31)] =
        (__bf16)proj_w[j];
  }
}

// ---------------- kernel 1: QKV GEMM, two windows per block -----------------
// 256 thr (8 waves = 8 M-tiles over M=128). A panel staged once (96KB LDS),
// per-wave A-fragments in VGPRs, B fragments streamed from L2-resident
// prepped weights as TWO independent accumulator chains per iteration.
__global__ __launch_bounds__(256) void swin_qkv(
    const float* __restrict__ x, const __bf16* __restrict__ pwQ,
    const float* __restrict__ qkv_b, __bf16* __restrict__ qws) {
  __shared__ __align__(16) __bf16 sA[128 * CDIM];   // 96 KB
  __shared__ int sRow[128];
  const int w0 = blockIdx.x * 2;
  const int tid = threadIdx.x;
  const int lane = tid & 31, wave = tid >> 5;       // wave = M-tile 0..7
  const int half = lane >> 4, l16 = lane & 15;

  if (tid < 128) {
    int r = tid & 63;
    sRow[tid] = (r < NTOK) ? winRowBase(w0 + (tid >> 6), r) : -1;
  }
  __syncthreads();

  for (int idx = tid; idx < 128 * (CDIM / 4); idx += 256) {
    int r = idx / (CDIM / 4), c4 = (idx % (CDIM / 4)) * 4;
    float4 v = make_float4(0.f, 0.f, 0.f, 0.f);
    if (sRow[r] >= 0) v = *(const float4*)&x[sRow[r] + c4];
    bf16x4 b;
    b[0] = (__bf16)v.x; b[1] = (__bf16)v.y; b[2] = (__bf16)v.z; b[3] = (__bf16)v.w;
    *(bf16x4*)&sA[r * CDIM + c4] = b;
  }
  __syncthreads();

  bf16x16 afr[KT];
#pragma unroll
  for (int kk = 0; kk < KT; ++kk)
    afr[kk] = fragA(&sA[(wave * 16 + l16) * CDIM + kk * 32], half);

  const int win = w0 + (wave >> 2);
  const int mt = wave & 3;
  const float scale = 0.17677669529663687f;  // 1/sqrt(32)

#pragma unroll 1
  for (int p = 0; p < 36; ++p) {            // 72 N-tiles in even/odd pairs
    const int nt = p * 2;                   // pair shares one head & segment
    const __bf16* b0 = pwQ + (size_t)nt * (KT * 512) + l16 * 32;
    const __bf16* b1 = b0 + (KT * 512);
    f32x8 acc0 = zero8(), acc1 = zero8();
#pragma unroll
    for (int kk = 0; kk < KT; ++kk) {
      bf16x16 f0 = fragB(b0 + kk * 512, half);
      bf16x16 f1 = fragB(b1 + kk * 512, half);
      acc0 = wmma_bf16(afr[kk], f0, acc0);
      acc1 = wmma_bf16(afr[kk], f1, acc1);
    }
    const int col0 = nt * 16;               // multiple of 32
    const int s = col0 / CDIM, rem = col0 - s * CDIM;
    const int h = rem >> 5;
    const float scl = (s == 0) ? scale : 1.0f;
    const float bias0 = qkv_b[col0 + l16];
    const float bias1 = qkv_b[col0 + 16 + l16];
    __bf16* dst =
        qws + ((size_t)((win * 3 + s) * NHEAD + h)) * (NTOK * HD) + l16;
#pragma unroll
    for (int e = 0; e < 8; ++e) {
      int row = mt * 16 + e + 8 * half;
      if (row < NTOK) {
        dst[row * HD]      = (__bf16)((acc0[e] + bias0) * scl);
        dst[row * HD + 16] = (__bf16)((acc1[e] + bias1) * scl);
      }
    }
  }
}

// ---------------- kernel 2: windowed attention ------------------------------
// grid(BW, NHEAD), 128 thr (4 waves). Q/K panels copied via async-to-LDS.
__global__ __launch_bounds__(128) void swin_attn(
    const __bf16* __restrict__ qws, const float* __restrict__ table,
    __bf16* __restrict__ ows) {
  __shared__ __align__(16) __bf16 sQ[64 * 32];
  __shared__ __align__(16) __bf16 sK[64 * 32];
  __shared__ __align__(16) __bf16 sVT[32 * 64];   // sVT[d*64 + m]
  __shared__ __align__(16) float  sS[64 * 64];
  __shared__ __align__(16) __bf16 sP[64 * 64];
  __shared__ int sCoord[64];
  __shared__ int sReg[64];

  const int w = blockIdx.x, h = blockIdx.y;
  const int tid = threadIdx.x;
  const int lane = tid & 31, wave = tid >> 5;
  const int half = lane >> 4, l16 = lane & 15;

  if (tid < 64) {
    int t = tid;
    if (t < NTOK) {
      int i = t / WSZ, j = t % WSZ;
      sCoord[t] = i * (2 * WSZ - 1) + j;
      int wi = w & 255, wh = wi >> 4, ww = wi & 15;
      int rh = (wh < NWS - 1) ? 0 : ((i < WSZ - SSZ) ? 1 : 2);
      int rw = (ww < NWS - 1) ? 0 : ((j < WSZ - SSZ) ? 1 : 2);
      sReg[t] = rh * 3 + rw;
    } else { sCoord[t] = 0; sReg[t] = -1; }
  }

  const __bf16* qg = qws + ((size_t)((w * 3 + 0) * NHEAD + h)) * (NTOK * HD);
  const __bf16* kg = qws + ((size_t)((w * 3 + 1) * NHEAD + h)) * (NTOK * HD);
  const __bf16* vg = qws + ((size_t)((w * 3 + 2) * NHEAD + h)) * (NTOK * HD);

  {  // async copies of the contiguous 49x32 bf16 Q/K panels (196 x 16B each)
    unsigned lq = (unsigned)(size_t)&sQ[0];
    unsigned lk = (unsigned)(size_t)&sK[0];
    for (int idx = tid; idx < (NTOK * HD) / 8; idx += 128) {
      async_ld16(lq + idx * 16, qg + idx * 8);
      async_ld16(lk + idx * 16, kg + idx * 8);
    }
  }
  for (int idx = tid; idx < (64 - NTOK) * HD; idx += 128) {
    sQ[NTOK * HD + idx] = (__bf16)0.0f;
    sK[NTOK * HD + idx] = (__bf16)0.0f;
  }
  for (int idx = tid; idx < 64 * 32; idx += 128) {
    int r = idx >> 5, c = idx & 31;
    sVT[c * 64 + r] = (r < NTOK) ? vg[r * HD + c] : (__bf16)0.0f;
  }
  wait_async0();
  __syncthreads();

  // ---- S = Q @ K^T ----
  f32x8 acc[4];
#pragma unroll
  for (int i = 0; i < 4; ++i) acc[i] = zero8();
  {
    bf16x16 a = fragA(&sQ[(wave * 16 + l16) * 32], half);
#pragma unroll
    for (int nt = 0; nt < 4; ++nt)
      acc[nt] = wmma_bf16(a, fragB(&sK[(nt * 16 + l16) * 32], half), acc[nt]);
  }
#pragma unroll
  for (int nt = 0; nt < 4; ++nt) {
#pragma unroll
    for (int e = 0; e < 8; ++e) {
      int row = wave * 16 + e + 8 * half;
      int col = nt * 16 + l16;
      float val = -1e30f;
      if (row < NTOK && col < NTOK) {
        int ridx = sCoord[row] + sCoord[48 - col];
        val = acc[nt][e] + table[ridx * NHEAD + h];
        if (sReg[row] != sReg[col]) val -= 100.0f;
      }
      sS[row * 64 + col] = val;
    }
  }
  __syncthreads();

  // ---- softmax: one thread per row; padded cols are -1e30 -> exp 0 ----
  if (tid < 64) {
    const float* rowp = &sS[tid * 64];
    float m = -1e30f;
    for (int c = 0; c < 64; ++c) m = fmaxf(m, rowp[c]);
    float sum = 0.0f;
    for (int c = 0; c < 64; ++c) sum += __expf(rowp[c] - m);
    float inv = 1.0f / sum;
    for (int c = 0; c < 64; ++c)
      sP[tid * 64 + c] = (__bf16)(__expf(rowp[c] - m) * inv);
  }
  __syncthreads();

  // ---- O = P @ V (K=64 in two steps) ----
  f32x8 o[2];
#pragma unroll
  for (int i = 0; i < 2; ++i) o[i] = zero8();
#pragma unroll
  for (int ks = 0; ks < 2; ++ks) {
    bf16x16 ap = fragA(&sP[(wave * 16 + l16) * 64 + ks * 32], half);
#pragma unroll
    for (int nt = 0; nt < 2; ++nt)
      o[nt] = wmma_bf16(ap, fragB(&sVT[(nt * 16 + l16) * 64 + ks * 32], half),
                        o[nt]);
  }
#pragma unroll
  for (int nt = 0; nt < 2; ++nt) {
#pragma unroll
    for (int e = 0; e < 8; ++e) {
      int row = wave * 16 + e + 8 * half;
      if (row < NTOK)
        ows[((size_t)(w * NTOK + row)) * CDIM + h * HD + nt * 16 + l16] =
            (__bf16)o[nt][e];
    }
  }
}

// ---------------- kernel 3: proj GEMM, two windows per block ----------------
// A panels are contiguous bf16 -> async-to-LDS copies. Scatter f32 rows
// through the inverse roll mapping (bijection, each element written once).
__global__ __launch_bounds__(256) void swin_proj(
    const __bf16* __restrict__ ows, const __bf16* __restrict__ pwP,
    const float* __restrict__ proj_b, float* __restrict__ out) {
  __shared__ __align__(16) __bf16 sA[128 * CDIM];   // 96 KB
  __shared__ int sDst[128];
  const int w0 = blockIdx.x * 2;
  const int tid = threadIdx.x;
  const int lane = tid & 31, wave = tid >> 5;
  const int half = lane >> 4, l16 = lane & 15;

  if (tid < 128) {
    int r = tid & 63;
    sDst[tid] = (r < NTOK) ? winRowBase(w0 + (tid >> 6), r) : -1;
  }

  {  // async copy: two contiguous 49x384 bf16 panels (2352 x 16B each)
    const __bf16* ag0 = ows + (size_t)w0 * (NTOK * CDIM);
    const __bf16* ag1 = ows + (size_t)(w0 + 1) * (NTOK * CDIM);
    unsigned la0 = (unsigned)(size_t)&sA[0];
    unsigned la1 = (unsigned)(size_t)&sA[64 * CDIM];
    for (int idx = tid; idx < (NTOK * CDIM) / 8; idx += 256) {
      async_ld16(la0 + idx * 16, ag0 + idx * 8);
      async_ld16(la1 + idx * 16, ag1 + idx * 8);
    }
  }
  for (int idx = tid; idx < (64 - NTOK) * CDIM / 4; idx += 256) {
    bf16x4 z; z[0] = z[1] = z[2] = z[3] = (__bf16)0.0f;
    *(bf16x4*)&sA[NTOK * CDIM + idx * 4] = z;
    *(bf16x4*)&sA[(64 + NTOK) * CDIM + idx * 4] = z;
  }
  wait_async0();
  __syncthreads();

  bf16x16 afr[KT];
#pragma unroll
  for (int kk = 0; kk < KT; ++kk)
    afr[kk] = fragA(&sA[(wave * 16 + l16) * CDIM + kk * 32], half);

  const int mt = wave & 3, wb = (wave >> 2) * 64;

#pragma unroll 1
  for (int p = 0; p < 12; ++p) {           // 24 N-tiles in pairs
    const int nt = p * 2;
    const __bf16* b0 = pwP + (size_t)nt * (KT * 512) + l16 * 32;
    const __bf16* b1 = b0 + (KT * 512);
    f32x8 acc0 = zero8(), acc1 = zero8();
#pragma unroll
    for (int kk = 0; kk < KT; ++kk) {
      bf16x16 f0 = fragB(b0 + kk * 512, half);
      bf16x16 f1 = fragB(b1 + kk * 512, half);
      acc0 = wmma_bf16(afr[kk], f0, acc0);
      acc1 = wmma_bf16(afr[kk], f1, acc1);
    }
    const int col = nt * 16 + l16;
    const float bias0 = proj_b[col];
    const float bias1 = proj_b[col + 16];
#pragma unroll
    for (int e = 0; e < 8; ++e) {
      int row = mt * 16 + e + 8 * half;
      if (row < NTOK) {
        int db = sDst[wb + row];
        out[db + col]      = acc0[e] + bias0;
        out[db + col + 16] = acc1[e] + bias1;
      }
    }
  }
}

// --------------------------------- launcher ---------------------------------
extern "C" void kernel_launch(void* const* d_in, const int* in_sizes, int n_in,
                              void* d_out, int out_size, void* d_ws,
                              size_t ws_size, hipStream_t stream) {
  (void)in_sizes; (void)n_in; (void)out_size; (void)ws_size;
  const float* x      = (const float*)d_in[0];
  const float* qkv_w  = (const float*)d_in[1];
  const float* qkv_b  = (const float*)d_in[2];
  const float* proj_w = (const float*)d_in[3];
  const float* proj_b = (const float*)d_in[4];
  const float* table  = (const float*)d_in[5];
  // d_in[6]=H, d_in[7]=W fixed at 112 -> compile-time constants.

  __bf16* qws = (__bf16*)d_ws;                                  // 231 MB
  __bf16* ows = qws + (size_t)BW * 3 * NHEAD * NTOK * HD;       //  77 MB
  __bf16* pwQ = ows + (size_t)BW * NTOK * CDIM;                 // 864 KB
  __bf16* pwP = pwQ + (size_t)CDIM * CQKV;                      // 288 KB
  float* out = (float*)d_out;

  const int prepTot = CDIM * CQKV + CDIM * CDIM;
  hipLaunchKernelGGL(prep_weights, dim3((prepTot + 255) / 256), dim3(256), 0,
                     stream, qkv_w, proj_w, pwQ, pwP);
  hipLaunchKernelGGL(swin_qkv, dim3(BW / 2), dim3(256), 0, stream,
                     x, pwQ, qkv_b, qws);
  hipLaunchKernelGGL(swin_attn, dim3(BW, NHEAD), dim3(128), 0, stream,
                     qws, table, ows);
  hipLaunchKernelGGL(swin_proj, dim3(BW / 2), dim3(256), 0, stream,
                     ows, pwP, proj_b, out);
}
